// LocationSensitiveAttention_62543313764738
// MI455X (gfx1250) — matrix operand order repaired and checked by reading, not compiled, not run on hardware
//
#include <hip/hip_runtime.h>
#include <hip/hip_bf16.h>

#define BB 32
#define TT 4096
#define QDIM 1024
#define MDIM 512
#define ADIM 128
#define FF 32
#define KK 31
#define PADC 15

#define NSPLIT 16
#define CHUNK (TT / NSPLIT)   // 256
#define TILE 16
#define NTILES (CHUNK / TILE) // 16

#define MROW 516              // padded fp32 LDS row stride (floats)
#define BROW 520              // padded bf16 LDS row stride (bf16 elems) -> conflict-free b128
#define LROW 40               // padded bf16 loc row stride

typedef __attribute__((ext_vector_type(16))) __bf16 v16bf;
typedef __attribute__((ext_vector_type(4)))  __bf16 v4bf;
typedef __attribute__((ext_vector_type(8)))  float  v8f;

__device__ __forceinline__ float fast_tanh(float x) {
  // 1 - 2/(exp(2x)+1): v_exp + v_rcp; exact at +-saturation
  const float e = __expf(2.0f * x);
  return 1.0f - 2.0f * __builtin_amdgcn_rcpf(e + 1.0f);
}

__device__ __forceinline__ void sched_fence() {
#if __has_builtin(__builtin_amdgcn_sched_barrier)
  __builtin_amdgcn_sched_barrier(0);   // cap scheduler-induced register pressure
#endif
}

// ---------------------------------------------------------------------------
// Prep 1: pqc[b,a] = query@Wq + bq + bm + bl + conv_b@Wl   (all additive-score
// constants folded into one per-(b,a) value; v_b dropped: softmax-invariant)
// ---------------------------------------------------------------------------
__global__ void prep_pqc(const float* __restrict__ q, const float* __restrict__ Wq,
                         const float* __restrict__ bq, const float* __restrict__ bm,
                         const float* __restrict__ bl, const float* __restrict__ convb,
                         const float* __restrict__ Wl, float* __restrict__ pqc) {
  const int b = blockIdx.x;
  const int a = threadIdx.x;                       // 128 threads
  float acc = bq[a] + bm[a] + bl[a];
  for (int f = 0; f < FF; ++f) acc += convb[f] * Wl[f * ADIM + a];
  const float* qb = q + b * QDIM;
  for (int i = 0; i < QDIM; ++i) acc += qb[i] * Wq[i * ADIM + a];
  pqc[b * ADIM + a] = acc;
}

// ---------------------------------------------------------------------------
// Prep 2: shuffle a [K, 128] fp32 weight into bf16 WMMA B-fragment order:
// dst[((kb*8+nb)*32+lane)*16+e] = W[kb*32 + (e>>3)*16 + (lane>=16)*8 + (e&7)][nb*16 + (lane&15)]
// so each lane loads its whole 16-element fragment as one contiguous 32B chunk.
// ---------------------------------------------------------------------------
__global__ void prep_frag(const float* __restrict__ W, __bf16* __restrict__ dst,
                          int kblocks) {
  const int tid  = blockIdx.x * 256 + threadIdx.x;
  const int e    = tid & 15;
  const int lane = (tid >> 4) & 31;
  const int nb   = (tid >> 9) & 7;
  const int kb   = tid >> 12;
  if (kb >= kblocks) return;
  const int k = kb * 32 + (e >> 3) * 16 + (lane >> 4) * 8 + (e & 7);
  const int n = nb * 16 + (lane & 15);
  dst[tid] = (__bf16)W[k * ADIM + n];
}

// ---------------------------------------------------------------------------
// Main fused kernel: per (split, batch) workgroup of 256 threads (8 waves):
//   stream 16x512 memory tiles (global->LDS once; fp32 copy for fp32 context
//   accumulation + single shared bf16 copy for WMMA A-fragments), score GEMM
//   via bf16 WMMA (serial per-wave chain; 4 resident waves/SIMD saturate the
//   matrix pipe across waves), location conv + loc GEMM (one extra WMMA),
//   fast tanh + v-dot, online softmax, context accumulation.
// ---------------------------------------------------------------------------
__global__ __launch_bounds__(256, 4)
void attn_main(const float* __restrict__ memory, const float* __restrict__ alignments,
               const unsigned char* __restrict__ mask, const float* __restrict__ conv_w,
               const float* __restrict__ v_w, const __bf16* __restrict__ wm_frag,
               const __bf16* __restrict__ wl_frag, const float* __restrict__ pqc,
               float* __restrict__ scores, float* __restrict__ part_m,
               float* __restrict__ part_s, float* __restrict__ part_c) {
  __shared__ __align__(16) float  tilef[TILE * MROW];        // 33 KB fp32 mem tile
  __shared__ __align__(16) __bf16 tileb[TILE * BROW];        // 16.6 KB bf16 mem tile
  __shared__ __align__(16) __bf16 locb[TILE * LROW];         // loc features (bf16)
  __shared__ float cw[FF * 2 * KK];                          // conv weights
  __shared__ float aw[(TILE + KK - 1) * 2];                  // alignment window
  __shared__ float sc[TILE];
  __shared__ float pp[TILE];
  __shared__ float st_m, st_s, st_fac;

  const int sp   = blockIdx.x;       // T split
  const int b    = blockIdx.y;       // batch row
  const int thr  = threadIdx.x;
  const int lane = thr & 31;
  const int wv   = thr >> 5;         // wave id 0..7 -> owns AD cols [16wv,16wv+16)

  for (int i = thr; i < FF * 2 * KK; i += 256) cw[i] = conv_w[i];

  // Persistent B-fragments: Wm column slab (16 k-blocks) + Wl (1 k-block)
  v16bf bfr[16];
#pragma unroll
  for (int kb = 0; kb < 16; ++kb)
    bfr[kb] = *(const v16bf*)(wm_frag + ((kb * 8 + wv) * 32 + lane) * 16);
  const v16bf wlb = *(const v16bf*)(wl_frag + (wv * 32 + lane) * 16);

  const int   acol   = wv * 16 + (lane & 15);
  const float vlane  = v_w[acol];
  const float pqlane = pqc[b * ADIM + acol];

  float c0 = 0.f, c1 = 0.f;                        // context accum, md = 2*thr(+1)
  const long t0base = (long)sp * CHUNK;
  const float* memb = memory + (long)b * TT * MDIM;

  if (thr == 0) { st_m = -__builtin_inff(); st_s = 0.f; }

  for (int tile = 0; tile < NTILES; ++tile) {
    const long t0 = t0base + tile * TILE;
    __syncthreads();                               // LDS tile reuse fence

    // ---- 1. global -> LDS (fp32 tile + bf16 tile, converted ONCE in regs).
    //         Two 4-float4 chunks with a scheduling fence between them to cap
    //         transient register pressure (loads are prefetch-warmed anyway).
    const float4* gsrc = (const float4*)(memb + t0 * MDIM);
#pragma unroll
    for (int g = 0; g < 2; ++g) {
#pragma unroll
      for (int j = g * 4; j < g * 4 + 4; ++j) {
        const int f4  = thr + j * 256;             // 2048 float4s total
        const int row = f4 >> 7;
        const int c4  = f4 & 127;
        const float4 v = gsrc[f4];
        *(float4*)&tilef[row * MROW + c4 * 4] = v;
        v4bf h;
        h[0] = (__bf16)v.x; h[1] = (__bf16)v.y;
        h[2] = (__bf16)v.z; h[3] = (__bf16)v.w;
        *(v4bf*)&tileb[row * BROW + c4 * 4] = h;
      }
      sched_fence();
    }
    if (tile + 1 < NTILES)
      __builtin_prefetch(memb + (t0 + TILE) * MDIM + thr * 32, 0, 0);

    // ---- 2. alignment window for conv (zero-padded at T edges)
    if (thr < (TILE + KK - 1) * 2) {
      const int  tr = thr >> 1, ch = thr & 1;
      const long tg = t0 - PADC + tr;
      float v = 0.f;
      if (tg >= 0 && tg < TT) v = alignments[((long)b * TT + tg) * 2 + ch];
      aw[tr * 2 + ch] = v;
    }
    if (thr < TILE) sc[thr] = 0.f;
    __syncthreads();

    // ---- 3. location conv -> bf16 loc tile [16 t x 32 f] (partial unroll to
    //         keep few cw/aw values live at once)
    {
      const int t  = thr >> 4;
      const int f0 = (thr & 15) * 2;
      float a0 = 0.f, a1 = 0.f;
      for (int ch = 0; ch < 2; ++ch) {
        const float* cw0 = &cw[(f0) * 2 * KK + ch * KK];
        const float* cw1 = &cw[(f0 + 1) * 2 * KK + ch * KK];
#pragma unroll 8
        for (int k = 0; k < KK; ++k) {
          const float av = aw[(t + k) * 2 + ch];
          a0 += av * cw0[k];
          a1 += av * cw1[k];
        }
      }
      locb[t * LROW + f0]     = (__bf16)a0;
      locb[t * LROW + f0 + 1] = (__bf16)a1;
    }
    __syncthreads();

    // ---- 4. WMMA: acc = loc @ Wl  (K = F = 32, exactly one bf16 WMMA)
    const int row = lane & 15, half = lane >> 4;
    v8f acc = {};
    {
      v16bf afr;
      ((float4*)&afr)[0] = *(const float4*)&locb[row * LROW + half * 8];
      ((float4*)&afr)[1] = *(const float4*)&locb[row * LROW + 16 + half * 8];
      acc = __builtin_amdgcn_wmma_f32_16x16x32_bf16(false, afr, false, wlb,
                                                    (short)0, acc, false, false);
    }

    // ---- 5. main GEMM: acc += mem_tile(bf16) @ Wm_slab, K = 512
    //         per k-block: 2x ds_load_b128 + 1 WMMA
    {
      const __bf16* srcb = &tileb[row * BROW + half * 8];
#pragma unroll
      for (int kb = 0; kb < 16; ++kb) {
        v16bf afr;
        ((float4*)&afr)[0] = *(const float4*)(srcb + kb * 32);       // K kb*32+half*8..
        ((float4*)&afr)[1] = *(const float4*)(srcb + kb * 32 + 16);  // K +16
        acc = __builtin_amdgcn_wmma_f32_16x16x32_bf16(false, afr, false, bfr[kb],
                                                      (short)0, acc, false, false);
      }
    }

    // ---- 6. score[t] = sum_a v[a]*tanh(pqc + acc): half-wave reduce + LDS add
    {
      float vals[8];
#pragma unroll
      for (int r = 0; r < 8; ++r) {
        float ctr = vlane * fast_tanh(pqlane + acc[r]);
#pragma unroll
        for (int mdx = 1; mdx < 16; mdx <<= 1) ctr += __shfl_xor(ctr, mdx, 32);
        vals[r] = ctr;
      }
      if ((lane & 15) == 0) {
        const int tb = (lane >> 4) * 8;      // lane0 -> t 0..7, lane16 -> t 8..15
#pragma unroll
        for (int r = 0; r < 8; ++r) atomicAdd(&sc[tb + r], vals[r]);
      }
    }
    __syncthreads();

    // ---- 7. wave 0: mask, persist raw scores, online-softmax update (parallel)
    if (wv == 0) {
      float s = -__builtin_inff();
      if (lane < TILE) {
        s = sc[lane];
        if (mask[(long)b * TT + t0 + lane]) s = -__builtin_inff();
        scores[(long)b * TT + t0 + lane] = s;
      }
      float tm = s;
#pragma unroll
      for (int mdx = 1; mdx < 16; mdx <<= 1) tm = fmaxf(tm, __shfl_xor(tm, mdx, 32));
      const float m_old = st_m;
      const float nm = fmaxf(m_old, tm);
      float p = 0.f;
      if (lane < TILE) { p = __expf(s - nm); pp[lane] = p; }
      float ps = p;
#pragma unroll
      for (int mdx = 1; mdx < 16; mdx <<= 1) ps += __shfl_xor(ps, mdx, 32);
      if (lane == 0) {
        const float fac = __expf(m_old - nm);
        st_s = st_s * fac + ps;
        st_m = nm;
        st_fac = fac;
      }
    }
    __syncthreads();

    // ---- 8. context accum from the SAME LDS tile (fp32; memory read once)
    {
      const float fac = st_fac;
      c0 *= fac; c1 *= fac;
      const float* col = &tilef[thr * 2];
#pragma unroll
      for (int t = 0; t < TILE; ++t) {
        const float p = pp[t];
        c0 += p * col[t * MROW];
        c1 += p * col[t * MROW + 1];
      }
    }
  }

  __syncthreads();
  if (thr == 0) {
    part_m[b * NSPLIT + sp] = st_m;
    part_s[b * NSPLIT + sp] = st_s;
  }
  float* pc = part_c + (long)(b * NSPLIT + sp) * MDIM;
  pc[2 * thr]     = c0;
  pc[2 * thr + 1] = c1;
}

// ---------------------------------------------------------------------------
// Combine split partials -> context output + global (m, 1/S) per batch row
// ---------------------------------------------------------------------------
__global__ void combine_ctx(const float* __restrict__ part_m, const float* __restrict__ part_s,
                            const float* __restrict__ part_c, float* __restrict__ ctx_out,
                            float* __restrict__ gm, float* __restrict__ gis) {
  const int b = blockIdx.x;
  const int thr = threadIdx.x;                     // 256
  __shared__ float wts[NSPLIT];
  __shared__ float ssum;
  if (thr == 0) {
    float m = -__builtin_inff();
    for (int j = 0; j < NSPLIT; ++j) m = fmaxf(m, part_m[b * NSPLIT + j]);
    float S = 0.f;
    for (int j = 0; j < NSPLIT; ++j) {
      const float w = __expf(part_m[b * NSPLIT + j] - m);
      wts[j] = w;
      S += w * part_s[b * NSPLIT + j];
    }
    ssum = S; gm[b] = m; gis[b] = 1.0f / S;
  }
  __syncthreads();
  const float inv = 1.0f / ssum;
  float a0 = 0.f, a1 = 0.f;
  for (int j = 0; j < NSPLIT; ++j) {
    const float* pc = part_c + (long)(b * NSPLIT + j) * MDIM;
    const float w = wts[j];
    a0 += w * pc[2 * thr];
    a1 += w * pc[2 * thr + 1];
  }
  ctx_out[(long)b * MDIM + 2 * thr]     = a0 * inv;
  ctx_out[(long)b * MDIM + 2 * thr + 1] = a1 * inv;
}

// ---------------------------------------------------------------------------
// alignment[b,t] = exp(score - m_b) * invS_b   (scores are L2-resident, 512 KB)
// ---------------------------------------------------------------------------
__global__ void align_out_k(const float* __restrict__ scores, const float* __restrict__ gm,
                            const float* __restrict__ gis, float* __restrict__ aout) {
  const int idx = blockIdx.x * 256 + threadIdx.x;  // B*T = 131072
  const int b = idx >> 12;                         // / 4096
  aout[idx] = __expf(scores[idx] - gm[b]) * gis[b];
}

extern "C" void kernel_launch(void* const* d_in, const int* in_sizes, int n_in,
                              void* d_out, int out_size, void* d_ws, size_t ws_size,
                              hipStream_t stream) {
  const float* query      = (const float*)d_in[0];
  const float* memory     = (const float*)d_in[1];
  const float* alignments = (const float*)d_in[2];
  const unsigned char* mask = (const unsigned char*)d_in[3];
  const float* Wq     = (const float*)d_in[4];
  const float* bq     = (const float*)d_in[5];
  const float* Wm     = (const float*)d_in[6];
  const float* bm     = (const float*)d_in[7];
  const float* conv_w = (const float*)d_in[8];
  const float* conv_b = (const float*)d_in[9];
  const float* Wl     = (const float*)d_in[10];
  const float* bl     = (const float*)d_in[11];
  const float* v_w    = (const float*)d_in[12];
  // d_in[13] (v_b) shifts all scores uniformly -> softmax-invariant, unused.

  char* ws = (char*)d_ws;
  __bf16* wm_frag = (__bf16*)(ws + 0);           // 512*128*2    = 131072
  __bf16* wl_frag = (__bf16*)(ws + 131072);      // 32*128*2     = 8192
  float*  pqc     = (float*)(ws + 139264);       // 32*128*4     = 16384
  float*  scores  = (float*)(ws + 155648);       // 32*4096*4    = 524288
  float*  part_m  = (float*)(ws + 679936);       // 32*16*4      = 2048
  float*  part_s  = (float*)(ws + 681984);       // 32*16*4      = 2048
  float*  part_c  = (float*)(ws + 684032);       // 32*16*512*4  = 1048576
  float*  gm      = (float*)(ws + 1732608);      // 32*4
  float*  gis     = (float*)(ws + 1732736);      // 32*4

  float* ctx_out = (float*)d_out;                // [B, 512]
  float* align_o = (float*)d_out + BB * MDIM;    // [B, 4096]

  prep_pqc<<<BB, ADIM, 0, stream>>>(query, Wq, bq, bm, bl, conv_b, Wl, pqc);
  prep_frag<<<256, 256, 0, stream>>>(Wm, wm_frag, 16);
  prep_frag<<<16, 256, 0, stream>>>(Wl, wl_frag, 1);
  attn_main<<<dim3(NSPLIT, BB), 256, 0, stream>>>(memory, alignments, mask, conv_w, v_w,
                                                  wm_frag, wl_frag, pqc, scores,
                                                  part_m, part_s, part_c);
  combine_ctx<<<BB, 256, 0, stream>>>(part_m, part_s, part_c, ctx_out, gm, gis);
  align_out_k<<<512, 256, 0, stream>>>(scores, gm, gis, align_o);
}
